// RINAttentionModel_75892072121103
// MI455X (gfx1250) — matrix operand order repaired and checked by reading, not compiled, not run on hardware
//
#include <hip/hip_runtime.h>
#include <math.h>

// ---------------- problem constants (match reference) ----------------
#define D_MODEL 1024
#define N_HEADS 16
#define D_HEAD  64
#define BATCH   8
#define SEQ     4096
#define NCHUNK  32                 // S-chunks for deterministic context partials
#define SCHUNK  (SEQ / NCHUNK)     // 128
#define PHI_F        1.6180339887498949f
#define INV_SCALE_F  0.08838834764831843f   // 1/sqrt(2*D_HEAD) = 1/sqrt(128)

typedef __attribute__((ext_vector_type(16))) _Float16 v16h;
typedef __attribute__((ext_vector_type(8)))  _Float16 v8h;
typedef __attribute__((ext_vector_type(8)))  float    v8f;

// ---------------- k1: theta_q -> (qc, qs), B*D elements ----------------
__global__ void k_theta_q(const float* __restrict__ x, const float* __restrict__ t,
                          const float* __restrict__ wq, const float* __restrict__ bq,
                          float* __restrict__ qc, float* __restrict__ qs) {
  int idx = blockIdx.x * blockDim.x + threadIdx.x;
  if (idx >= BATCH * D_MODEL) return;
  int b = idx / D_MODEL, d = idx % D_MODEL;
  float th = x[idx] / (1.0f + fabsf(wq[d])) + bq[d] + t[b] * PHI_F;
  float sn, cs;
  __sincosf(th, &sn, &cs);
  qc[idx] = cs;
  qs[idx] = sn;
}

// ---------------- k2: fused scores + softmax, one WG per (b,h) ----------------
// Streams the head slice of cached_states once, does sincos inline, block-wide
// softmax over S=4096 (16 values per thread, 256 threads). This is the
// roofline-critical kernel: 134 MB HBM stream + 67M sincos.
__global__ void k_scores_softmax(const float* __restrict__ cached,
                                 const float* __restrict__ qc, const float* __restrict__ qs,
                                 const float* __restrict__ wk, const float* __restrict__ bk,
                                 float* __restrict__ weights) {
  const int b = blockIdx.x / N_HEADS;
  const int h = blockIdx.x % N_HEADS;
  const int tid = threadIdx.x;  // 256 threads

  __shared__ float s_qc[D_HEAD], s_qs[D_HEAD], s_inv[D_HEAD], s_bk[D_HEAD];
  __shared__ float red[256];

  if (tid < D_HEAD) {
    int dp = h * D_HEAD + tid;
    s_qc[tid]  = qc[b * D_MODEL + dp];
    s_qs[tid]  = qs[b * D_MODEL + dp];
    s_inv[tid] = 1.0f / (1.0f + fabsf(wk[dp]));
    s_bk[tid]  = bk[dp];
  }
  __syncthreads();

  float sc[16];
  for (int i = 0; i < 16; ++i) {
    const int s = tid + i * 256;
    const float* row = cached + ((size_t)b * SEQ + s) * D_MODEL + h * D_HEAD;
    __builtin_prefetch(row + 256 * D_MODEL, 0, 0);   // next i-iteration's row
    float acc = 0.0f;
#pragma unroll 8
    for (int d = 0; d < D_HEAD; ++d) {
      float th = fmaf(row[d], s_inv[d], s_bk[d]);
      float sn, cs;
      __sincosf(th, &sn, &cs);
      acc = fmaf(s_qc[d], cs, fmaf(s_qs[d], sn, acc));
    }
    sc[i] = acc * INV_SCALE_F;
  }

  // block max (deterministic tree)
  float mx = sc[0];
#pragma unroll
  for (int i = 1; i < 16; ++i) mx = fmaxf(mx, sc[i]);
  red[tid] = mx; __syncthreads();
  for (int o = 128; o > 0; o >>= 1) {
    if (tid < o) red[tid] = fmaxf(red[tid], red[tid + o]);
    __syncthreads();
  }
  mx = red[0]; __syncthreads();

  // exp + block sum
  float sum = 0.0f;
#pragma unroll
  for (int i = 0; i < 16; ++i) { sc[i] = __expf(sc[i] - mx); sum += sc[i]; }
  red[tid] = sum; __syncthreads();
  for (int o = 128; o > 0; o >>= 1) {
    if (tid < o) red[tid] += red[tid + o];
    __syncthreads();
  }
  const float invd = 1.0f / red[0];

  float* wrow = weights + ((size_t)b * N_HEADS + h) * SEQ;
  for (int i = 0; i < 16; ++i) wrow[tid + i * 256] = sc[i] * invd;
}

// ---------------- k3: collapse heads: wsum[b,s] = sum_h weights ----------------
__global__ void k_wsum(const float* __restrict__ weights, float* __restrict__ wsum) {
  int idx = blockIdx.x * blockDim.x + threadIdx.x;
  if (idx >= BATCH * SEQ) return;
  int b = idx / SEQ, s = idx % SEQ;
  const float* p = weights + (size_t)b * N_HEADS * SEQ + s;
  float acc = 0.0f;
#pragma unroll
  for (int hh = 0; hh < N_HEADS; ++hh) acc += p[(size_t)hh * SEQ];
  wsum[idx] = acc;
}

// ---------------- k4: context partials over S chunks (2nd cached pass, L2-hot) ----
__global__ void k_ctx_partial(const float* __restrict__ cached, const float* __restrict__ wsum,
                              float* __restrict__ part) {
  const int b  = blockIdx.x / NCHUNK;
  const int c  = blockIdx.x % NCHUNK;
  const int m4 = threadIdx.x * 4;       // 256 threads * float4 = 1024 cols
  float4 acc = make_float4(0.f, 0.f, 0.f, 0.f);
  const int s0 = c * SCHUNK;
  for (int s = s0; s < s0 + SCHUNK; ++s) {
    const float w = wsum[b * SEQ + s];
    const float4 v = *(const float4*)(cached + ((size_t)b * SEQ + s) * D_MODEL + m4);
    __builtin_prefetch(cached + ((size_t)b * SEQ + s + 4) * D_MODEL + m4, 0, 0);
    acc.x = fmaf(w, v.x, acc.x);
    acc.y = fmaf(w, v.y, acc.y);
    acc.z = fmaf(w, v.z, acc.z);
    acc.w = fmaf(w, v.w, acc.w);
  }
  float* o = part + ((size_t)b * NCHUNK + c) * D_MODEL + m4;
  o[0] = acc.x; o[1] = acc.y; o[2] = acc.z; o[3] = acc.w;
}

// ---------------- k5: reduce partials -> zero-padded f16 context (16 x D) -------
// Rows 8..15 are written as zero so the WMMA kernel needs no lane predication.
__global__ void k_ctx_reduce_f16(const float* __restrict__ part, _Float16* __restrict__ ctx16h) {
  int idx = blockIdx.x * blockDim.x + threadIdx.x;
  if (idx >= 16 * D_MODEL) return;
  int b = idx / D_MODEL, m = idx % D_MODEL;
  float acc = 0.0f;
  if (b < BATCH) {
#pragma unroll
    for (int c = 0; c < NCHUNK; ++c) acc += part[((size_t)b * NCHUNK + c) * D_MODEL + m];
  }
  ctx16h[idx] = (_Float16)acc;
}

// ---------------- k5b: convert w_out to f16 (same row-major layout) ----------------
__global__ void k_cvt_wout(const float* __restrict__ w_out, _Float16* __restrict__ w_out_h) {
  int i4 = (blockIdx.x * blockDim.x + threadIdx.x) * 4;
  if (i4 >= D_MODEL * D_MODEL) return;
  const float4 v = *(const float4*)(w_out + i4);
  w_out_h[i4 + 0] = (_Float16)v.x;
  w_out_h[i4 + 1] = (_Float16)v.y;
  w_out_h[i4 + 2] = (_Float16)v.z;
  w_out_h[i4 + 3] = (_Float16)v.w;
}

// ---------------- k6: out = x + context @ w_out^T via v_wmma_f32_16x16x32_f16 ----
// D[16x16 tile] = A(16xK) x B(Kx16): A = padded f16 context, B[k][n] = w_out[n][k].
// One wave per 16-wide n-tile; 64 waves cover N=1024. Per the CDNA5 16-bit VGPR
// layouts, each lane's A fragment is two contiguous 8-half runs:
//   a[0..7]  = A[m][kb + half*8      .. +8)      (one 16B load)
//   a[8..15] = A[m][kb + 16 + half*8 .. +8)      (one 16B load)
// and each lane's B fragment uses the same k order from one w_out^T column:
//   b[0..7]  = w_out_h[n*D + kb + half*8      .. +8)  (one 16B load)
//   b[8..15] = w_out_h[n*D + kb + 16 + half*8 .. +8)  (one 16B load)
__global__ void k_out_wmma(const _Float16* __restrict__ ctx16h,
                           const _Float16* __restrict__ w_out_h,
                           const float* __restrict__ x, float* __restrict__ out) {
  const int lane  = threadIdx.x & 31;
  const int wave  = blockIdx.x * (blockDim.x >> 5) + (threadIdx.x >> 5);
  const int ncol0 = wave * 16;          // n-tile base, 64 waves -> 1024 cols
  const int mrow  = lane & 15;          // A row / B column within tile
  const int half  = lane >> 4;

  const _Float16* arow = ctx16h + mrow * D_MODEL;
  const _Float16* bcol = w_out_h + (size_t)(ncol0 + mrow) * D_MODEL;

  v8f acc = {};
  for (int kb = 0; kb < D_MODEL; kb += 32) {
    const v8h alo = *(const v8h*)(arow + kb + half * 8);
    const v8h ahi = *(const v8h*)(arow + kb + 16 + half * 8);
    const v16h a  = __builtin_shufflevector(alo, ahi,
                      0, 1, 2, 3, 4, 5, 6, 7, 8, 9, 10, 11, 12, 13, 14, 15);
    const v8h blo = *(const v8h*)(bcol + kb + half * 8);
    const v8h bhi = *(const v8h*)(bcol + kb + 16 + half * 8);
    const v16h bm = __builtin_shufflevector(blo, bhi,
                      0, 1, 2, 3, 4, 5, 6, 7, 8, 9, 10, 11, 12, 13, 14, 15);
    acc = __builtin_amdgcn_wmma_f32_16x16x32_f16(
        /*neg_a=*/false, a, /*neg_b=*/false, bm,
        /*c_mod=*/(short)0, acc, /*reuse_a=*/false, /*reuse_b=*/false);
  }

  // D rows: VGPR j = M (half*8 + j); lanes 0-15 carry rows 0..7 = real batches
  if (half == 0) {
    const int n = ncol0 + mrow;
#pragma unroll
    for (int j = 0; j < 8; ++j)
      out[j * D_MODEL + n] = x[j * D_MODEL + n] + acc[j];
  }
}

// ---------------- launch ----------------
extern "C" void kernel_launch(void* const* d_in, const int* in_sizes, int n_in,
                              void* d_out, int out_size, void* d_ws, size_t ws_size,
                              hipStream_t stream) {
  (void)in_sizes; (void)n_in; (void)out_size; (void)ws_size;
  const float* x      = (const float*)d_in[0];
  const float* cached = (const float*)d_in[1];
  const float* t      = (const float*)d_in[2];
  const float* wq     = (const float*)d_in[3];
  const float* bq     = (const float*)d_in[4];
  const float* wk     = (const float*)d_in[5];
  const float* bk     = (const float*)d_in[6];
  const float* w_out  = (const float*)d_in[7];
  float* out = (float*)d_out;

  // workspace layout: f32 region first, then f16 region (all 32B-aligned:
  // every size below is a multiple of 8192 floats / big powers of two).
  float* ws      = (float*)d_ws;
  float* qc      = ws;                                        // B*D
  float* qs      = qc + BATCH * D_MODEL;                      // B*D
  float* weights = qs + BATCH * D_MODEL;                      // B*H*S
  float* wsum    = weights + (size_t)BATCH * N_HEADS * SEQ;   // B*S
  float* part    = wsum + BATCH * SEQ;                        // B*NCHUNK*D
  _Float16* ctx16h  = (_Float16*)(part + (size_t)BATCH * NCHUNK * D_MODEL); // 16*D halfs
  _Float16* w_out_h = ctx16h + 16 * D_MODEL;                  // D*D halfs (2 MB)

  k_theta_q       <<<(BATCH * D_MODEL + 255) / 256,     256, 0, stream>>>(x, t, wq, bq, qc, qs);
  k_scores_softmax<<<BATCH * N_HEADS,                   256, 0, stream>>>(cached, qc, qs, wk, bk, weights);
  k_wsum          <<<(BATCH * SEQ + 255) / 256,         256, 0, stream>>>(weights, wsum);
  k_ctx_partial   <<<BATCH * NCHUNK,                    256, 0, stream>>>(cached, wsum, part);
  k_ctx_reduce_f16<<<(16 * D_MODEL + 255) / 256,        256, 0, stream>>>(part, ctx16h);
  k_cvt_wout      <<<(D_MODEL * D_MODEL / 4 + 255) / 256, 256, 0, stream>>>(w_out, w_out_h);
  k_out_wmma      <<<8,                                 256, 0, stream>>>(ctx16h, w_out_h, x, out);
}
